// RIM_12214886989947
// MI455X (gfx1250) — matrix operand order repaired and checked by reading, not compiled, not run on hardware
//
#include <hip/hip_runtime.h>
#include <hip/hip_bf16.h>

// ---------------------------------------------------------------------------
// RIM forward (B=64, T=196, U=8, H=256) for gfx1250 (MI455X).
//  K1: pre-swizzle fp32 weights into WMMA B-operand records (f16), init hs.
//  K2: persistent recurrent kernel; all activations kept in LDS in WMMA
//      A-operand record layout -> GEMM inner loop = ds_load_b128 (A) +
//      global_load_b128 (B, L2-resident) + v_wmma_f32_16x16x32_f16.
//  K3: MLP head.
// Workspace: ~6.6 MB.
// ---------------------------------------------------------------------------

typedef _Float16 half_t;
typedef __attribute__((ext_vector_type(16))) _Float16 v16h;
typedef __attribute__((ext_vector_type(8)))  _Float16 v8h;
typedef __attribute__((ext_vector_type(8)))  float    v8f;

#define Bn   64
#define Tn   196
#define Un   8
#define Hn   256
#define AKn  4
#define DINn 64
#define KIn  64
#define HCn  2
#define KCn  32
#define VCn  256
#define UHn  (Un * Hn)        // 2048
#define HCVC (HCn * VCn)      // 512

#define BT      16            // batch tile per workgroup
#define NWAVES  8             // 256 threads

// ---- workspace layout (bytes): f16 weights in B-record layout ----
#define WS_HS    0u                     // 131072 f32  (B*U*H state master)
#define WS_WQ    524288u                // 131072 f16
#define WS_WIN   786432u                // 131072 f16
#define WS_WREC  1048576u               // 524288 f16
#define WS_WCK   2097152u               // 131072 f16
#define WS_WCQ   2359296u               // 131072 f16
#define WS_WCV   2621440u               // 1048576 f16
#define WS_WCO   4718592u               // 1048576 f16

// ---- LDS layout (bytes) ----
#define SM_HS16 0                       // 32768 f16 A-records: old hs      (65536)
#define SM_H16  65536                   // 32768 f16 A-records: h_new       (65536)
#define SM_VCA  131072                  // 65536 f16 A-records: vc -> o     (131072)
#define SM_QBUF SM_VCA                  // alias: 16*8*64 f32 q             (32768)
#define SM_ATTX (SM_VCA + 32768)        // alias: attx A-records 8192 f16   (16384)
#define SM_KC   262144                  // 16*512 f16                       (16384)
#define SM_QC   278528                  // 16*512 f16                       (16384)
#define SM_K    294912                  // 16*2*64 f32                      (8192)
#define SM_V    303104                  // 16*2*64 f32                      (8192)
#define SM_ATT  311296                  // 16*8*2 f32                       (1024)
#define SM_MASK 312320                  // 16*8 f32                         (512)
#define SM_XT   312832                  // 16 f32                           (64)
#define SM_TOTAL 312896

// ---------------------------------------------------------------------------
__device__ __forceinline__ v8f wmma16(v16h a, v16h b, v8f c) {
  return __builtin_amdgcn_wmma_f32_16x16x32_f16(false, a, false, b,
                                                (short)0, c, false, false);
}

// One 512-half record = one full WMMA operand tile (A 16x32 or B 32x16):
// record[lane*16 + i].  Load = two b128s, already in register layout.
__device__ __forceinline__ v16h ld_rec(const half_t* base, int rec) {
  const int lane = threadIdx.x & 31;
  return *(const v16h*)(base + ((size_t)rec * 32 + lane) * 16);
}

// Scalar slot of element (k,row) inside an A-record region (per ISA 16-bit
// A 16x32 layout): lanes<16 hold K {0..7,16..23}, lanes>=16 hold the rest.
__device__ __forceinline__ int aslot(int k, int row) {
  const int kt   = k >> 5;
  const int kw   = k & 31;
  const int lane = row + ((kw & 8) << 1);          // +16 if kw in [8,16)|[24,32)
  const int i    = (kw & 7) | ((kw & 16) >> 1);    // +8 if kw >= 16
  return (kt * 32 + lane) * 16 + i;
}

// Native CDNA5 tanh (trans op; pad the hazard window explicitly).
__device__ __forceinline__ float fast_tanh(float x) {
  float r;
  asm volatile("v_tanh_f32 %0, %1\n\tv_nop\n\tv_nop" : "=v"(r) : "v"(x));
  return r;
}

// ---------------------------------------------------------------------------
// K1: swizzle weight (U x K x N, fp32) into B-records:
// dst[(((u*NT+nt)*KT+kt)*32 + lane)*16 + i], lane = col + 16*khalf,
// element k = kt*32 + khalf*8 + (i&7) + 16*(i>>3), n = nt*16 + col.
// ---------------------------------------------------------------------------
__device__ __forceinline__ void convertB(const float* __restrict__ src,
                                         half_t* __restrict__ dst,
                                         int Uc, int Kc, int Nc, float scale,
                                         long tid, long stride) {
  const int NT = Nc / 16, KT = Kc / 32;
  const long total = (long)Uc * Kc * Nc;
  for (long o = tid; o < total; o += stride) {
    long r = o;
    const int i    = (int)(r & 15); r >>= 4;
    const int lane = (int)(r & 31); r >>= 5;
    const int kt   = (int)(r % KT); r /= KT;
    const int nt   = (int)(r % NT); r /= NT;
    const int u    = (int)r;
    const int n = nt * 16 + (lane & 15);
    const int k = kt * 32 + (lane >> 4) * 8 + (i & 7) + ((i >> 3) << 4);
    dst[o] = (_Float16)(src[((long)u * Kc + k) * Nc + n] * scale);
  }
}

__global__ __launch_bounds__(256) void rim_prep_kernel(
    const float* __restrict__ wq,  const float* __restrict__ win,
    const float* __restrict__ wrec,const float* __restrict__ wck,
    const float* __restrict__ wcq, const float* __restrict__ wcv,
    const float* __restrict__ wco,
    half_t* __restrict__ wqB,  half_t* __restrict__ winB,
    half_t* __restrict__ wrecB,half_t* __restrict__ wckB,
    half_t* __restrict__ wcqB, half_t* __restrict__ wcvB,
    half_t* __restrict__ wcoB, float* __restrict__ hs) {
  const long tid = (long)blockIdx.x * blockDim.x + threadIdx.x;
  const long stride = (long)gridDim.x * blockDim.x;
  const float QSC = 0.17677669529663689f;   // 1/sqrt(KC)
  convertB(wq,   wqB,   Un, Hn,   KIn,  1.0f, tid, stride);
  convertB(win,  winB,  Un, DINn, Hn,   1.0f, tid, stride);
  convertB(wrec, wrecB, Un, Hn,   Hn,   1.0f, tid, stride);
  convertB(wck,  wckB,  Un, Hn,   64,   1.0f, tid, stride);
  convertB(wcq,  wcqB,  Un, Hn,   64,   QSC,  tid, stride);
  convertB(wcv,  wcvB,  Un, Hn,   HCVC, 1.0f, tid, stride);
  convertB(wco,  wcoB,  Un, HCVC, VCn,  1.0f, tid, stride);
  for (long i = tid; i < Bn * UHn; i += stride) hs[i] = 0.001f;
}

// ---------------------------------------------------------------------------
// K2: full T=196 recurrence; 4 workgroups x 16 batch rows.
// ---------------------------------------------------------------------------
__global__ __launch_bounds__(256) void rim_scan_kernel(
    const float* __restrict__ x,   const float* __restrict__ wk,
    const float* __restrict__ bk,  const float* __restrict__ wv,
    const float* __restrict__ bv,  const float* __restrict__ brec,
    const half_t* __restrict__ wqB,  const half_t* __restrict__ winB,
    const half_t* __restrict__ wrecB,const half_t* __restrict__ wckB,
    const half_t* __restrict__ wcqB, const half_t* __restrict__ wcvB,
    const half_t* __restrict__ wcoB, float* __restrict__ hs) {
  __shared__ __align__(32) unsigned char smem[SM_TOTAL];
  half_t* hs16  = (half_t*)(smem + SM_HS16);  // per-u 4096 halves (8 K-recs)
  half_t* h16   = (half_t*)(smem + SM_H16);   // per-u 4096 halves
  half_t* vcA   = (half_t*)(smem + SM_VCA);   // per-u 8192 halves (16 K-recs)
  float*  qbuf  = (float*)(smem + SM_QBUF);   // [b][u*64+n]   (alias vcA)
  half_t* attxA = (half_t*)(smem + SM_ATTX);  // per-u 1024 halves (alias vcA)
  half_t* kc16  = (half_t*)(smem + SM_KC);    // [b][flat 512]
  half_t* qc16  = (half_t*)(smem + SM_QC);
  float*  kbuf  = (float*)(smem + SM_K);
  float*  vbuf  = (float*)(smem + SM_V);
  float*  attb  = (float*)(smem + SM_ATT);
  float*  maskb = (float*)(smem + SM_MASK);
  float*  xtb   = (float*)(smem + SM_XT);

  const int tid   = threadIdx.x;
  const int wave  = tid >> 5;
  const int lane  = tid & 31;
  const int ccol  = lane & 15;
  const int crow0 = (lane >> 4) << 3;
  const int bbase = blockIdx.x * BT;
  float* hsg = hs + (size_t)bbase * UHn;

  // init f16 shadow of hs
  for (int i = tid; i < BT * UHn; i += 256) hs16[i] = (_Float16)0.001f;
  __syncthreads();

  for (int t = 0; t < Tn; ++t) {
    // ---- P1: xt, input k/v rows --------------------------------------
    if (tid < BT) xtb[tid] = x[(size_t)(bbase + tid) * Tn + t];
    __syncthreads();
    for (int i = tid; i < BT * DINn; i += 256) {
      const int b = i >> 6, j = i & 63;
      const float xv = xtb[b];
      kbuf[b * 128 + j]      = xv * wk[j] + bk[j];
      kbuf[b * 128 + 64 + j] = bk[j];
      vbuf[b * 128 + j]      = xv * wv[j] + bv[j];
      vbuf[b * 128 + 64 + j] = bv[j];
    }
    // ---- P2: q = hs @ wq_in  (per unit 16x64, K=256) -----------------
    for (int tile = wave; tile < Un * 4; tile += NWAVES) {
      const int u = tile >> 2, nt = tile & 3;
      v8f c;
#pragma unroll
      for (int r = 0; r < 8; ++r) c[r] = 0.0f;
      const half_t* Bw = wqB + (size_t)(u * 4 + nt) * 8 * 512;
#pragma unroll
      for (int kt = 0; kt < 8; ++kt)
        c = wmma16(ld_rec(hs16 + u * 4096, kt), ld_rec(Bw, kt), c);
#pragma unroll
      for (int r = 0; r < 8; ++r)
        qbuf[(crow0 + r) * (Un * KIn) + u * KIn + nt * 16 + ccol] = c[r];
    }
    __syncthreads();
    // ---- P3: input attention (softmax over n=2) ----------------------
    if (tid < BT * Un) {
      const int b = tid >> 3, u = tid & 7;
      const float* q = qbuf + b * (Un * KIn) + u * KIn;
      float d0 = 0.f, d1 = 0.f;
      for (int j = 0; j < KIn; ++j) {
        d0 += q[j] * kbuf[b * 128 + j];
        d1 += q[j] * kbuf[b * 128 + 64 + j];
      }
      d0 *= 0.125f; d1 *= 0.125f;
      const float mx = fmaxf(d0, d1);
      const float e0 = __expf(d0 - mx), e1 = __expf(d1 - mx);
      const float inv = 1.0f / (e0 + e1);
      attb[(b * Un + u) * 2 + 0] = e0 * inv;
      attb[(b * Un + u) * 2 + 1] = e1 * inv;
    }
    __syncthreads();
    // ---- P4: top-AK units by smallest null score ---------------------
    if (tid < BT) {
      float nullv[Un], m[Un]; bool used[Un];
#pragma unroll
      for (int u = 0; u < Un; ++u) {
        nullv[u] = attb[(tid * Un + u) * 2 + 1];
        m[u] = 0.f; used[u] = false;
      }
      for (int s = 0; s < AKn; ++s) {
        int best = 0; float bv2 = 1e30f;
#pragma unroll
        for (int u = 0; u < Un; ++u)
          if (!used[u] && nullv[u] < bv2) { bv2 = nullv[u]; best = u; }
        used[best] = true; m[best] = 1.f;
      }
#pragma unroll
      for (int u = 0; u < Un; ++u) maskb[tid * Un + u] = m[u];
    }
    __syncthreads();
    // ---- P5: attx = mask * (att @ v) -> A-records --------------------
    for (int i = tid; i < BT * Un * DINn; i += 256) {
      const int b = i >> 9, rest = i & 511, u = rest >> 6, j = rest & 63;
      const float a0 = attb[(b * Un + u) * 2 + 0];
      const float a1 = attb[(b * Un + u) * 2 + 1];
      const float mv = maskb[b * Un + u];
      attxA[u * 1024 + aslot(j, b)] =
          (_Float16)(mv * (a0 * vbuf[b * 128 + j] + a1 * vbuf[b * 128 + 64 + j]));
    }
    __syncthreads();
    // ---- P6: pre = attx@win + hs@wrec + brec ; h_new -> h16 ----------
    for (int tile = wave; tile < Un * 16; tile += NWAVES) {
      const int u = tile >> 4, nt = tile & 15;
      const int n0 = nt << 4;
      const float bias = brec[u * Hn + n0 + ccol];
      v8f c;
#pragma unroll
      for (int r = 0; r < 8; ++r) c[r] = bias;
      const half_t* Bi = winB + (size_t)((u * 16 + nt) * 2) * 512;
#pragma unroll
      for (int kt = 0; kt < 2; ++kt)
        c = wmma16(ld_rec(attxA + u * 1024, kt), ld_rec(Bi, kt), c);
      const half_t* Br = wrecB + (size_t)((u * 16 + nt) * 8) * 512;
      __builtin_prefetch(Br, 0, 1);
#pragma unroll
      for (int kt = 0; kt < 8; ++kt)
        c = wmma16(ld_rec(hs16 + u * 4096, kt), ld_rec(Br, kt), c);
#pragma unroll
      for (int r = 0; r < 8; ++r) {
        const int brow = crow0 + r;
        const float hold = hsg[brow * UHn + u * Hn + n0 + ccol];
        h16[u * 4096 + aslot(n0 + ccol, brow)] =
            (_Float16)(0.5f * hold + 0.5f * fast_tanh(c[r]));   // LEAKY=0.5
      }
    }
    __syncthreads();
    // ---- P7: kc = h@wck, qc = h@wcq (scale folded) -------------------
    for (int tile = wave; tile < 64; tile += NWAVES) {
      const int which = tile >> 5;
      const int u = (tile >> 2) & 7, nt = tile & 3;
      const half_t* Bw = (which ? wcqB : wckB) + (size_t)((u * 4 + nt) * 8) * 512;
      v8f c;
#pragma unroll
      for (int r = 0; r < 8; ++r) c[r] = 0.0f;
#pragma unroll
      for (int kt = 0; kt < 8; ++kt)
        c = wmma16(ld_rec(h16 + u * 4096, kt), ld_rec(Bw, kt), c);
      half_t* dst = which ? qc16 : kc16;
#pragma unroll
      for (int r = 0; r < 8; ++r)
        dst[(crow0 + r) * 512 + u * 64 + (nt << 4) + ccol] = (_Float16)c[r];
    }
    // ---- P8: vc = h@wcv (per unit 16x512) -> A-records ---------------
    for (int tile = wave; tile < Un * 32; tile += NWAVES) {
      const int u = tile >> 5, nt = tile & 31;
      const int n0 = nt << 4;
      v8f c;
#pragma unroll
      for (int r = 0; r < 8; ++r) c[r] = 0.0f;
      const half_t* Bw = wcvB + (size_t)((u * 32 + nt) * 8) * 512;
      __builtin_prefetch(Bw, 0, 1);
#pragma unroll
      for (int kt = 0; kt < 8; ++kt)
        c = wmma16(ld_rec(h16 + u * 4096, kt), ld_rec(Bw, kt), c);
#pragma unroll
      for (int r = 0; r < 8; ++r)
        vcA[u * 8192 + aslot(n0 + ccol, crow0 + r)] = (_Float16)c[r];
    }
    __syncthreads();
    // ---- P9: per-head attention; o overwrites vc in place (x ALPHA) --
    {
      const int b = tid >> 4, hc = (tid >> 3) & 1, u2 = tid & 7;
      float s[Un];
      const half_t* qrow = qc16 + b * 512 + hc * 256 + u2 * KCn;
#pragma unroll
      for (int v = 0; v < Un; ++v) {
        const half_t* krow = kc16 + b * 512 + hc * 256 + v * KCn;
        float acc = 0.f;
#pragma unroll
        for (int k = 0; k < KCn; ++k) acc += (float)qrow[k] * (float)krow[k];
        s[v] = acc;
      }
      float mx = s[0];
#pragma unroll
      for (int v = 1; v < Un; ++v) mx = fmaxf(mx, s[v]);
      float den = 0.f;
#pragma unroll
      for (int v = 0; v < Un; ++v) { s[v] = __expf(s[v] - mx); den += s[v]; }
      const float inv = 1.0f / den;
#pragma unroll
      for (int v = 0; v < Un; ++v) s[v] *= inv;
      // all 8 u2-lanes of a (b,hc) group live in one wave: lockstep
      // read-then-write keeps the in-place update race-free.
      for (int j0 = 0; j0 < VCn; j0 += 8) {
        float acc[8];
#pragma unroll
        for (int e = 0; e < 8; ++e) acc[e] = 0.f;
#pragma unroll
        for (int v = 0; v < Un; ++v) {
          const int uu = hc * 4 + (v >> 1);
          const int kk = ((v & 1) << 8) + j0;
          const v8h xv = *(const v8h*)(vcA + uu * 8192 + aslot(kk, b));
#pragma unroll
          for (int e = 0; e < 8; ++e) acc[e] += s[v] * (float)xv[e];
        }
        __builtin_amdgcn_wave_barrier();
        asm volatile("s_wait_dscnt 0" ::: "memory");
        v8h o;
#pragma unroll
        for (int e = 0; e < 8; ++e) o[e] = (_Float16)(0.5f * acc[e]);  // ALPHA
        const int uo = hc * 4 + (u2 >> 1);
        const int ko = ((u2 & 1) << 8) + j0;
        *(v8h*)(vcA + uo * 8192 + aslot(ko, b)) = o;
        __builtin_amdgcn_wave_barrier();
      }
    }
    __syncthreads();
    // ---- P10: h_comm = (0.5*o)@wco + 0.5*h_new; masked state update --
    for (int tile = wave; tile < Un * 16; tile += NWAVES) {
      const int u = tile >> 4, nt = tile & 15;
      const int n0 = nt << 4;
      v8f c;
#pragma unroll
      for (int r = 0; r < 8; ++r)
        c[r] = 0.5f * (float)h16[u * 4096 + aslot(n0 + ccol, crow0 + r)];
      const half_t* Bw = wcoB + (size_t)((u * 16 + nt) * 16) * 512;
      __builtin_prefetch(Bw, 0, 1);
#pragma unroll
      for (int kt = 0; kt < 16; ++kt)
        c = wmma16(ld_rec(vcA + u * 8192, kt), ld_rec(Bw, kt), c);
#pragma unroll
      for (int r = 0; r < 8; ++r) {
        const int brow = crow0 + r;
        if (maskb[brow * Un + u] > 0.5f) {
          hsg[brow * UHn + u * Hn + n0 + ccol] = c[r];
          hs16[u * 4096 + aslot(n0 + ccol, brow)] = (_Float16)c[r];
        }
      }
    }
    __syncthreads();
  }
}

// ---------------------------------------------------------------------------
// K3: out = relu(hs_flat @ w1 + b1) @ w2 + b2
// ---------------------------------------------------------------------------
__global__ __launch_bounds__(256) void rim_mlp_kernel(
    const float* __restrict__ hs, const float* __restrict__ w1,
    const float* __restrict__ b1, const float* __restrict__ w2,
    const float* __restrict__ b2, float* __restrict__ out) {
  __shared__ float hid[Bn * 200];
  const int tid = threadIdx.x;
  for (int p = tid; p < Bn * 200; p += 256) {
    const int b = p / 200, j = p % 200;
    const float* hr = hs + (size_t)b * UHn;
    float acc = b1[j];
    for (int k = 0; k < UHn; ++k) acc += hr[k] * w1[(size_t)k * 200 + j];
    hid[p] = fmaxf(acc, 0.0f);
  }
  __syncthreads();
  for (int p = tid; p < Bn * 10; p += 256) {
    const int b = p / 10, o = p % 10;
    float acc = b2[o];
    for (int k = 0; k < 200; ++k) acc += hid[b * 200 + k] * w2[k * 10 + o];
    out[p] = acc;
  }
}

// ---------------------------------------------------------------------------
extern "C" void kernel_launch(void* const* d_in, const int* in_sizes, int n_in,
                              void* d_out, int out_size, void* d_ws, size_t ws_size,
                              hipStream_t stream) {
  (void)in_sizes; (void)n_in; (void)out_size; (void)ws_size;
  const float* x    = (const float*)d_in[0];
  const float* wk   = (const float*)d_in[1];
  const float* bk   = (const float*)d_in[2];
  const float* wv   = (const float*)d_in[3];
  const float* bv   = (const float*)d_in[4];
  const float* wq   = (const float*)d_in[5];
  const float* win  = (const float*)d_in[6];
  const float* wrec = (const float*)d_in[7];
  const float* brec = (const float*)d_in[8];
  const float* wck  = (const float*)d_in[9];
  const float* wcq  = (const float*)d_in[10];
  const float* wcv  = (const float*)d_in[11];
  const float* wco  = (const float*)d_in[12];
  const float* w1   = (const float*)d_in[13];
  const float* b1   = (const float*)d_in[14];
  const float* w2   = (const float*)d_in[15];
  const float* b2   = (const float*)d_in[16];

  char* ws = (char*)d_ws;
  float*  hs    = (float*)(ws + WS_HS);
  half_t* wqB   = (half_t*)(ws + WS_WQ);
  half_t* winB  = (half_t*)(ws + WS_WIN);
  half_t* wrecB = (half_t*)(ws + WS_WREC);
  half_t* wckB  = (half_t*)(ws + WS_WCK);
  half_t* wcqB  = (half_t*)(ws + WS_WCQ);
  half_t* wcvB  = (half_t*)(ws + WS_WCV);
  half_t* wcoB  = (half_t*)(ws + WS_WCO);

  rim_prep_kernel<<<1024, 256, 0, stream>>>(
      wq, win, wrec, wck, wcq, wcv, wco,
      wqB, winB, wrecB, wckB, wcqB, wcvB, wcoB, hs);

  rim_scan_kernel<<<Bn / BT, 256, 0, stream>>>(
      x, wk, bk, wv, bv, brec,
      wqB, winB, wrecB, wckB, wcqB, wcvB, wcoB, hs);

  rim_mlp_kernel<<<1, 256, 0, stream>>>(hs, w1, b1, w2, b2, (float*)d_out);
}
